// LSTM_Dec_v2_31112743092808
// MI455X (gfx1250) — compile-verified, matrix operand order checked by reading, not run on hardware
//
#include <hip/hip_runtime.h>

typedef __attribute__((ext_vector_type(16))) __bf16 v16bf;
typedef __attribute__((ext_vector_type(8)))  float  v8f;

#define BATCH 256
#define HID   1024
#define IN    512
#define GATES 4096
#define TLEN  256

__device__ __forceinline__ float sigmoid_f(float x) {
    return 1.0f / (1.0f + __expf(-x));
}
__device__ __forceinline__ float tanh_f(float x) {
    float ax = fabsf(x);
    float e  = __expf(-2.0f * ax);          // in (0,1], no overflow
    float t  = (1.0f - e) / (1.0f + e);
    return copysignf(t, x);
}

__device__ __forceinline__ void wait_asynccnt0() {
#if __has_builtin(__builtin_amdgcn_s_wait_asynccnt)
    __builtin_amdgcn_s_wait_asynccnt(0);
#else
    asm volatile("s_wait_asynccnt 0x0" ::: "memory");
#endif
}

// Async copy of one 16-byte chunk global -> LDS, tracked by ASYNCcnt.
// ldsaddr = low 32 bits of the generic pointer.
__device__ __forceinline__ void async_copy_b128(unsigned ldsaddr, const void* gaddr) {
    asm volatile("global_load_async_to_lds_b128 %0, %1, off"
                 :: "v"(ldsaddr), "v"(gaddr)
                 : "memory");
}

// ---------------------------------------------------------------------------
// x_proj = ctx @ w_ih^T + b_ih + b_hh, stored in WMMA C-fragment layout:
//   xp[((bt*256 + nt)*8 + v)*32 + lane], element (M,N) of tile (bt,nt) maps
//   to lane = N%16 + 16*(M/8), v = M%8.
// ---------------------------------------------------------------------------
__global__ __launch_bounds__(256) void xproj_kernel(
    const float* __restrict__ ctx, const float* __restrict__ w_ih,
    const float* __restrict__ b_ih, const float* __restrict__ b_hh,
    float* __restrict__ xp) {
    int tid  = blockIdx.x * 256 + threadIdx.x;
    int lane = tid & 31;
    int v    = (tid >> 5) & 7;
    int nt   = (tid >> 8) & 255;
    int bt   = tid >> 16;
    int b = bt * 16 + v + 8 * (lane >> 4);
    int n = nt * 16 + (lane & 15);
    const float4* c4 = (const float4*)(ctx + (size_t)b * IN);
    const float4* w4 = (const float4*)(w_ih + (size_t)n * IN);
    float acc = 0.0f;
#pragma unroll 4
    for (int k = 0; k < IN / 4; ++k) {
        float4 a = c4[k], w = w4[k];
        acc += a.x * w.x + a.y * w.y + a.z * w.z + a.w * w.w;
    }
    xp[tid] = acc + b_ih[n] + b_hh[n];
}

// ---------------------------------------------------------------------------
// Pack w_hh (4096x1024 f32; row n = gate idx, col k = hidden) into bf16 WMMA
// B-tiles, lane-major: tile (nt,kt) at (nt*32+kt)*512, lane L pos p holds
// element (N = nt*16 + L%16, K = kt*32 + (p%8) + 8*(L/16) + 16*(p/8)).
// For a fixed jt, each gate's 32 K-tiles are contiguous (16384 bf16 = 32 KB).
// ---------------------------------------------------------------------------
__global__ __launch_bounds__(256) void packw_kernel(
    const float* __restrict__ w_hh, __bf16* __restrict__ Wpk) {
    int tid  = blockIdx.x * 256 + threadIdx.x;      // 4,194,304 total
    int p    = tid & 15;
    int lane = (tid >> 4) & 31;
    int kt   = (tid >> 9) & 31;
    int nt   = tid >> 14;
    int n = nt * 16 + (lane & 15);
    int k = kt * 32 + (p & 7) + 8 * (lane >> 4) + 16 * (p >> 3);
    Wpk[tid] = (__bf16)w_hh[(size_t)n * HID + k];
}

// Zero initial hidden state (packed A-layout) and cell state (fragment layout)
__global__ __launch_bounds__(256) void init_state_kernel(
    __bf16* __restrict__ hA, float* __restrict__ cbuf) {
    int tid = blockIdx.x * 256 + threadIdx.x;       // 262,144 total
    hA[tid]   = (__bf16)0.0f;
    cbuf[tid] = 0.0f;
}

// ---------------------------------------------------------------------------
// One LSTM timestep.  Grid: 64 blocks x 512 threads (16 waves).
//   block <-> jt (hidden tile 0..63), wave <-> bt (batch tile 0..15).
// The block's full weight slice (4 gates x 16 cols x K=1024, bf16 = 128 KB)
// is copied into LDS via GLOBAL_LOAD_ASYNC_TO_LDS_B128 (ASYNCcnt), then the
// K-loop runs barrier-free: per kt, 1 A-tile load + 4 LDS fragments + 4 WMMAs.
// ---------------------------------------------------------------------------
__global__ __launch_bounds__(512) void lstm_step_kernel(
    const float* __restrict__ xp, const __bf16* __restrict__ Wpk,
    const __bf16* __restrict__ hprev, __bf16* __restrict__ hnext,
    float* __restrict__ cbuf, float* __restrict__ out, int t) {
    __shared__ __align__(32) __bf16 Wlds[4 * 16384];   // 128 KB

    int tid  = threadIdx.x;
    int lane = tid & 31;
    int bt   = tid >> 5;          // wave id = batch tile
    int jt   = blockIdx.x;        // hidden tile

    // ---- async weight preload: 4 contiguous 32 KB streams -> LDS ----
    // Per gate: 2048 x 16B chunks; 512 threads x 4 chunks each.
    unsigned lds_base = (unsigned)(size_t)(&Wlds[0]);
#pragma unroll
    for (int g = 0; g < 4; ++g) {
        const uint4* src = (const uint4*)(Wpk + (size_t)(g * 64 + jt) * 16384);
        unsigned     dst = lds_base + (unsigned)(g * 32768);
#pragma unroll
        for (int i = 0; i < 4; ++i) {
            int c16 = i * 512 + tid;                 // uint4 index, 0..2047
            async_copy_b128(dst + (unsigned)(c16 * 16), (const void*)(src + c16));
        }
    }
    wait_asynccnt0();
    __syncthreads();

    v8f acc[4];
#pragma unroll
    for (int g = 0; g < 4; ++g) acc[g] = (v8f){};

    // ---- barrier-free K loop: 32 x (A load + 4 fragments + 4 WMMAs) ----
    for (int kt = 0; kt < 32; ++kt) {
        v16bf a = *(const v16bf*)(hprev + (size_t)(bt * 32 + kt) * 512 +
                                  lane * 16);
#pragma unroll
        for (int g = 0; g < 4; ++g) {
            v16bf bm = *(const v16bf*)(&Wlds[g * 16384 + kt * 512 + lane * 16]);
            acc[g] = __builtin_amdgcn_wmma_f32_16x16x32_bf16(
                false, a, false, bm, (short)0, acc[g], false, false);
        }
    }

    // ---- elementwise tail: gates -> (c,h) updates ----
    int m_hi = (lane >> 4) * 8;
    int nl   = lane & 15;
    int j    = jt * 16 + nl;          // global hidden index
    int kt2  = j >> 5;
    int k32  = j & 31;
    int lane2 = 16 * ((k32 >> 3) & 1);
    int p2    = (k32 & 7) + 8 * (k32 >> 4);

#pragma unroll
    for (int v = 0; v < 8; ++v) {
        int m = v + m_hi;             // batch index within tile
        float gi = acc[0][v] + xp[((bt * 256 + (  0 + jt)) * 8 + v) * 32 + lane];
        float gf = acc[1][v] + xp[((bt * 256 + ( 64 + jt)) * 8 + v) * 32 + lane];
        float gg = acc[2][v] + xp[((bt * 256 + (128 + jt)) * 8 + v) * 32 + lane];
        float go = acc[3][v] + xp[((bt * 256 + (192 + jt)) * 8 + v) * 32 + lane];

        int   cidx   = ((bt * 64 + jt) * 8 + v) * 32 + lane;
        float c_prev = cbuf[cidx];
        float i_g = sigmoid_f(gi);
        float f_g = sigmoid_f(gf);
        float g_g = tanh_f(gg);
        float o_g = sigmoid_f(go);
        float c_new = f_g * c_prev + i_g * g_g;
        float h_new = o_g * tanh_f(c_new);
        cbuf[cidx] = c_new;

        int b = bt * 16 + m;
        out[((size_t)b * TLEN + t) * HID + j] = h_new;
        // scatter h as bf16 into the next step's WMMA A-tile layout
        hnext[(size_t)(bt * 32 + kt2) * 512 + (m + lane2) * 16 + p2] =
            (__bf16)h_new;
    }
}

// ---------------------------------------------------------------------------
extern "C" void kernel_launch(void* const* d_in, const int* in_sizes, int n_in,
                              void* d_out, int out_size, void* d_ws,
                              size_t ws_size, hipStream_t stream) {
    (void)in_sizes; (void)n_in; (void)out_size; (void)ws_size;
    const float* ctx  = (const float*)d_in[0];  // 256 x 512
    const float* w_ih = (const float*)d_in[1];  // 4096 x 512
    const float* w_hh = (const float*)d_in[2];  // 4096 x 1024
    const float* b_ih = (const float*)d_in[3];  // 4096
    const float* b_hh = (const float*)d_in[4];  // 4096
    float* out = (float*)d_out;                 // 256 x 256 x 1024

    char* ws = (char*)d_ws;
    float*  xp   = (float*)(ws);                               // 4 MB
    __bf16* Wpk  = (__bf16*)(ws + (4u  << 20));                // 8 MB
    __bf16* hA0  = (__bf16*)(ws + (12u << 20));                // 512 KB
    __bf16* hA1  = (__bf16*)(ws + (12u << 20) + (512u << 10)); // 512 KB
    float*  cbuf = (float*)(ws + (13u << 20));                 // 1 MB

    hipLaunchKernelGGL(xproj_kernel, dim3(4096), dim3(256), 0, stream,
                       ctx, w_ih, b_ih, b_hh, xp);
    hipLaunchKernelGGL(packw_kernel, dim3(16384), dim3(256), 0, stream,
                       w_hh, Wpk);
    hipLaunchKernelGGL(init_state_kernel, dim3(1024), dim3(256), 0, stream,
                       hA0, cbuf);

    for (int t = 0; t < TLEN; ++t) {
        const __bf16* hp = (t & 1) ? hA1 : hA0;
        __bf16*       hn = (t & 1) ? hA0 : hA1;
        hipLaunchKernelGGL(lstm_step_kernel, dim3(64), dim3(512), 0, stream,
                           xp, Wpk, hp, hn, cbuf, out, t);
    }
}
